// CrossAttentionS2Audio_49847390437705
// MI455X (gfx1250) — matrix operand order repaired and checked by reading, not compiled, not run on hardware
//
#include <hip/hip_runtime.h>
#include <hip/hip_bf16.h>

typedef __attribute__((ext_vector_type(16))) __bf16 v16bf;
typedef __attribute__((ext_vector_type(8)))  float  v8f;

union FragBF {            // 16 bf16 = 32 bytes = two uint4 chunks (8 elems each)
    v16bf v;
    uint4 q[2];
    unsigned short u[16];
};
union U16x16 {
    uint4 q[2];
    unsigned short u[16];
};

__device__ __forceinline__ unsigned short f2bf_u(float f) {
    unsigned u = __float_as_uint(f);
    u = (u + 0x7FFFu + ((u >> 16) & 1u)) >> 16;   // round-to-nearest-even
    return (unsigned short)u;
}
__device__ __forceinline__ float bf2f(unsigned short s) {
    return __uint_as_float(((unsigned)s) << 16);
}
__device__ __forceinline__ v8f wmma_bf16(v16bf a, v16bf b, v8f c) {
    return __builtin_amdgcn_wmma_f32_16x16x32_bf16(false, a, false, b, (short)0, c, false, false);
}

// ---------------------------------------------------------------- constants
#define BB   8
#define TT   8
#define DD   768
#define HH   12
#define HDD  64
#define NTOK 1568          /* AP*T == NP*T */
#define MROWS 12544        /* B*NTOK */
// Q pre-scale folds softmax scale AND log2(e): exp(S-m) == exp2(S'-m')
#define QSCALE 0.1803368801111204f   /* 64^-0.5 * log2(e) */

// LDS strides (padded, 16B-aligned, conflict-free for b128 frag reads)
#define SKA 40    /* gemm A tile row stride  (80B  -> 20-bank step)  */
#define SKB 80    /* gemm B tile row stride  (160B -> 40-bank step)  */
#define SKV 72    /* attn K/V row stride     (144B -> 36-bank step)  */
#define SQT 120   /* attn Q^T row stride     (240B -> 60-bank step)  */

// ---------------------------------------------------------------- fp32 -> bf16
__global__ void cvt_bf16_kernel(const float* __restrict__ src,
                                unsigned short* __restrict__ dst, int n) {
    int i = (blockIdx.x * blockDim.x + threadIdx.x) * 4;
    if (i >= n) return;
    float4 v = *(const float4*)(src + i);
    ushort4 o;
    o.x = f2bf_u(v.x); o.y = f2bf_u(v.y); o.z = f2bf_u(v.z); o.w = f2bf_u(v.w);
    *(ushort4*)(dst + i) = o;
}

// ------------------------------------------------- pos-add + cast (= _add_pos)
__global__ void add_pos_cast_kernel(const float* __restrict__ src,
                                    const float* __restrict__ space,
                                    const float* __restrict__ temp,
                                    unsigned short* __restrict__ dst) {
    size_t i = ((size_t)blockIdx.x * blockDim.x + threadIdx.x) * 4;
    if (i >= (size_t)MROWS * DD) return;
    int d = (int)(i % DD);
    size_t row = i / DD;
    int b = (int)(row / NTOK);
    int n = (int)(row % NTOK);
    int p = n >> 3, t = n & 7;
    float4 sv = *(const float4*)(src + ((size_t)p * 64 + b * 8 + t) * DD + d);
    float4 sp = *(const float4*)(space + (size_t)p * DD + d);
    float4 tp = *(const float4*)(temp + (size_t)t * DD + d);
    ushort4 o;
    o.x = f2bf_u(sv.x + sp.x + tp.x);
    o.y = f2bf_u(sv.y + sp.y + tp.y);
    o.z = f2bf_u(sv.z + sp.z + tp.z);
    o.w = f2bf_u(sv.w + sp.w + tp.w);
    *(ushort4*)(dst + i) = o;
}

// ---------------------------------------------------------------- GEMM (WMMA)
// C[M,N] = A[M,K](bf16) @ W[N,K]^T(bf16) + bias
// MODE 0: out = bf16, row-major M x N
// MODE 1: out = fp32 scattered into d_out layout (final projection)
template <int MODE>
__global__ __launch_bounds__(256) void gemm_bf16_kernel(
        const unsigned short* __restrict__ A,
        const unsigned short* __restrict__ W,
        const float* __restrict__ bias,
        void* __restrict__ out, int M, int N, int K) {
    __shared__ unsigned short sA[128 * SKA];
    __shared__ unsigned short sB[32 * SKB];     // K-major: sB[k][n]

    int tid = threadIdx.x;
    int wave = tid >> 5, lane = tid & 31;
    int wm = wave & 3, wn = wave >> 2;          // 4 x 2 wave grid
    int ln = lane & 15, hw = lane >> 4;
    int mbase = blockIdx.x * 128, nbase = blockIdx.y * 64;

    v8f acc[2][2] = {};

    for (int kb = 0; kb < K; kb += 32) {
        __syncthreads();
        {   // A tile: 128x32, 16 bf16 per thread
            int row = tid >> 1, ch = tid & 1;
            const unsigned short* g = A + (size_t)(mbase + row) * K + kb + ch * 16;
            unsigned short* s = sA + row * SKA + ch * 16;
            *(uint4*)(s)     = *(const uint4*)(g);
            *(uint4*)(s + 8) = *(const uint4*)(g + 8);
        }
        if (tid < 128) {  // W tile 64x32 -> stored transposed sB[k][n]
            int n = tid >> 1, ch = tid & 1;
            U16x16 wv;
            const unsigned short* g = W + (size_t)(nbase + n) * K + kb + ch * 16;
            wv.q[0] = *(const uint4*)(g);
            wv.q[1] = *(const uint4*)(g + 8);
#pragma unroll
            for (int j = 0; j < 16; ++j) sB[(ch * 16 + j) * SKB + n] = wv.u[j];
        }
        __syncthreads();

        FragBF af[2], bf[2];
#pragma unroll
        for (int mt = 0; mt < 2; ++mt) {        // A layout: two 16B chunks @ 8*hw
            const unsigned short* s = sA + (wm * 32 + mt * 16 + ln) * SKA;
            af[mt].q[0] = *(const uint4*)(s + 8 * hw);
            af[mt].q[1] = *(const uint4*)(s + 16 + 8 * hw);
        }
#pragma unroll
        for (int nt = 0; nt < 2; ++nt) {        // B layout: lane->n, elems->k
            const unsigned short* s = sB + wn * 32 + nt * 16 + ln;
#pragma unroll
            for (int j = 0; j < 16; ++j)
                bf[nt].u[j] = s[(8 * hw + (j & 7) + 16 * (j >> 3)) * SKB];
        }
#pragma unroll
        for (int mt = 0; mt < 2; ++mt)
#pragma unroll
            for (int nt = 0; nt < 2; ++nt)
                acc[mt][nt] = wmma_bf16(af[mt].v, bf[nt].v, acc[mt][nt]);
    }

    float bv0 = bias[nbase + wn * 32 + ln];
    float bv1 = bias[nbase + wn * 32 + 16 + ln];
#pragma unroll
    for (int mt = 0; mt < 2; ++mt)
#pragma unroll
        for (int nt = 0; nt < 2; ++nt)
#pragma unroll
            for (int r = 0; r < 8; ++r) {
                int m = mbase + wm * 32 + mt * 16 + r + 8 * hw;
                int n = nbase + wn * 32 + nt * 16 + ln;
                float cv = acc[mt][nt][r] + (nt ? bv1 : bv0);
                if (MODE == 0) {
                    ((unsigned short*)out)[(size_t)m * N + n] = f2bf_u(cv);
                } else {
                    int b = m / NTOK, nn = m % NTOK;
                    int p = nn >> 3, t = nn & 7;
                    size_t dst = ((size_t)((2 + p) * 64 + b * 8 + t)) * DD + n;
                    ((float*)out)[dst] = cv;
                }
            }
}

// ---------------------------------------------------------- flash attention
// S^T = K.Q^T  (C-layout of S^T == A-layout of P  -> shuffle-free P.V)
// One block step over NKT*16 keys (NKT = 4 main, 2 tail).
template <int NKT>
__device__ __forceinline__ void attn_block(
        const unsigned short* sK, const unsigned short* sV,
        const FragBF* bq, v8f* oacc, float& m_old, float& lsum,
        int ln, int hw) {
    v8f st[NKT];
#pragma unroll
    for (int kt = 0; kt < NKT; ++kt) { v8f z = {}; st[kt] = z; }
#pragma unroll
    for (int s = 0; s < 2; ++s)
#pragma unroll
        for (int kt = 0; kt < NKT; ++kt) {
            FragBF a;
            const unsigned short* r = sK + (kt * 16 + ln) * SKV + s * 32 + 8 * hw;
            a.q[0] = *(const uint4*)(r);
            a.q[1] = *(const uint4*)(r + 16);
            st[kt] = wmma_bf16(a.v, bq[s].v, st[kt]);
        }

    // online softmax: lane owns q-row ln; holds NKT*8 of NKT*16 keys in-lane
    float mb = st[0][0];
#pragma unroll
    for (int kt = 0; kt < NKT; ++kt)
#pragma unroll
        for (int r = 0; r < 8; ++r) mb = fmaxf(mb, st[kt][r]);
    mb = fmaxf(mb, __shfl_xor(mb, 16, 32));
    float mn = fmaxf(m_old, mb);

    float ls = 0.0f;
    FragBF pf[NKT / 2];
#pragma unroll
    for (int kt = 0; kt < NKT; ++kt)
#pragma unroll
        for (int r = 0; r < 8; ++r) {
            float p = __builtin_amdgcn_exp2f(st[kt][r] - mn);  // arg <= 0: raw v_exp ok
            ls += p;
            pf[kt >> 1].v[(kt & 1) * 8 + r] = (__bf16)p;
        }
    ls += __shfl_xor(ls, 16, 32);
    float alpha = __builtin_amdgcn_exp2f(m_old - mn);
    lsum = lsum * alpha + ls;
    m_old = mn;

#pragma unroll
    for (int r = 0; r < 8; ++r) {               // per-row O rescale
        float ar = __shfl(alpha, r + 8 * hw, 32);
        oacc[0][r] *= ar; oacc[1][r] *= ar; oacc[2][r] *= ar; oacc[3][r] *= ar;
    }

#pragma unroll
    for (int s2 = 0; s2 < NKT / 2; ++s2)        // O += P.V
#pragma unroll
        for (int c = 0; c < 4; ++c) {
            FragBF bv;
            const unsigned short* p = sV + (s2 * 32) * SKV + c * 16 + ln;
#pragma unroll
            for (int j = 0; j < 16; ++j)
                bv.u[j] = p[(8 * hw + (j & 7) + 16 * (j >> 3)) * SKV];
            oacc[c] = wmma_bf16(pf[s2].v, bv.v, oacc[c]);
        }
}

__global__ __launch_bounds__(224) void attn_kernel(
        const unsigned short* __restrict__ Q,    // 12544 x 768 bf16
        const unsigned short* __restrict__ KV,   // 12544 x 1536 bf16 (k | v)
        unsigned short* __restrict__ O) {        // 12544 x 768 bf16
    __shared__ unsigned short sK[64 * SKV];
    __shared__ unsigned short sV[64 * SKV];
    __shared__ unsigned short sQT[64 * SQT];     // sQT[d][qrow_in_block]

    int tid = threadIdx.x, wave = tid >> 5, lane = tid & 31;
    int ln = lane & 15, hw = lane >> 4;
    int b = blockIdx.x / HH, h = blockIdx.x % HH;
    int qbase = blockIdx.y * 112;

    {   // stage Q^T, pre-scaled by SCALE*log2(e)
        int qr = ln, d0 = hw * 32;
        const unsigned short* g =
            Q + ((size_t)(b * NTOK + qbase + wave * 16 + qr)) * DD + h * HDD + d0;
        U16x16 t0, t1;
        t0.q[0] = *(const uint4*)(g);      t0.q[1] = *(const uint4*)(g + 8);
        t1.q[0] = *(const uint4*)(g + 16); t1.q[1] = *(const uint4*)(g + 24);
#pragma unroll
        for (int j = 0; j < 16; ++j)
            sQT[(d0 + j) * SQT + wave * 16 + qr] = f2bf_u(QSCALE * bf2f(t0.u[j]));
#pragma unroll
        for (int j = 0; j < 16; ++j)
            sQT[(d0 + 16 + j) * SQT + wave * 16 + qr] = f2bf_u(QSCALE * bf2f(t1.u[j]));
    }
    __syncthreads();

    FragBF bq[2];                                // Q^T B-fragments (fixed)
#pragma unroll
    for (int s = 0; s < 2; ++s) {
        const unsigned short* p = sQT + (s * 32) * SQT + wave * 16 + ln;
#pragma unroll
        for (int j = 0; j < 16; ++j)
            bq[s].u[j] = p[(8 * hw + (j & 7) + 16 * (j >> 3)) * SQT];
    }

    float m_old = -3.0e38f, lsum = 0.0f;
    v8f oacc[4] = {};

    const size_t kvrow = (size_t)b * NTOK * (2 * DD) + h * HDD;

    for (int kb = 0; kb < 24; ++kb) {            // 24 x 64 keys
        __syncthreads();
        {   // K tile: threads 0..127; V tile: threads 96..223 (128 tasks each)
            int taskK = tid, taskV = tid - 96;
            if (taskK < 128) {
                int key = taskK >> 1, off = (taskK & 1) * 32;
                const unsigned short* g = KV + kvrow + (size_t)(kb * 64 + key) * (2 * DD) + off;
                unsigned short* d = sK + key * SKV + off;
#pragma unroll
                for (int j = 0; j < 4; ++j) *(uint4*)(d + 8 * j) = *(const uint4*)(g + 8 * j);
            }
            if (taskV >= 0) {
                int key = taskV >> 1, off = (taskV & 1) * 32;
                const unsigned short* g = KV + kvrow + (size_t)(kb * 64 + key) * (2 * DD) + DD + off;
                unsigned short* d = sV + key * SKV + off;
#pragma unroll
                for (int j = 0; j < 4; ++j) *(uint4*)(d + 8 * j) = *(const uint4*)(g + 8 * j);
            }
        }
        __syncthreads();
        attn_block<4>(sK, sV, bq, oacc, m_old, lsum, ln, hw);
    }

    // tail: 32 keys (1536..1567)
    __syncthreads();
    {
        if (tid < 64) {
            int key = tid >> 1, off = (tid & 1) * 32;
            const unsigned short* g = KV + kvrow + (size_t)(1536 + key) * (2 * DD) + off;
            unsigned short* d = sK + key * SKV + off;
#pragma unroll
            for (int j = 0; j < 4; ++j) *(uint4*)(d + 8 * j) = *(const uint4*)(g + 8 * j);
        } else if (tid < 128) {
            int t2 = tid - 64;
            int key = t2 >> 1, off = (t2 & 1) * 32;
            const unsigned short* g = KV + kvrow + (size_t)(1536 + key) * (2 * DD) + DD + off;
            unsigned short* d = sV + key * SKV + off;
#pragma unroll
            for (int j = 0; j < 4; ++j) *(uint4*)(d + 8 * j) = *(const uint4*)(g + 8 * j);
        }
    }
    __syncthreads();
    attn_block<2>(sK, sV, bq, oacc, m_old, lsum, ln, hw);

    // epilogue: divide by row-sum, store bf16
#pragma unroll
    for (int r = 0; r < 8; ++r) {
        float lr = __shfl(lsum, r + 8 * hw, 32);
        float inv = __builtin_amdgcn_rcpf(lr);
        int qrow = qbase + wave * 16 + r + 8 * hw;
        size_t base = ((size_t)(b * NTOK + qrow)) * DD + h * HDD + ln;
#pragma unroll
        for (int c = 0; c < 4; ++c)
            O[base + c * 16] = f2bf_u(oacc[c][r] * inv);
    }
}

// ---------------------------------------------------------------- cls copy
__global__ void copy_cls_kernel(const float* __restrict__ audio,
                                float* __restrict__ out) {
    int i = (blockIdx.x * blockDim.x + threadIdx.x) * 4;
    if (i >= 2 * 64 * DD) return;
    *(float4*)(out + i) = *(const float4*)(audio + i);
}

// ---------------------------------------------------------------- launch
extern "C" void kernel_launch(void* const* d_in, const int* in_sizes, int n_in,
                              void* d_out, int out_size, void* d_ws, size_t ws_size,
                              hipStream_t stream) {
    const float* s_x        = (const float*)d_in[0];
    const float* audio      = (const float*)d_in[1];
    const float* clip_sp    = (const float*)d_in[2];
    const float* audio_sp   = (const float*)d_in[3];
    const float* clip_tp    = (const float*)d_in[4];
    const float* audio_tp   = (const float*)d_in[5];
    const float* q_w        = (const float*)d_in[6];
    const float* q_b        = (const float*)d_in[7];
    const float* kv_w       = (const float*)d_in[8];
    const float* kv_b       = (const float*)d_in[9];
    const float* proj_w     = (const float*)d_in[10];
    const float* proj_b     = (const float*)d_in[11];
    float* out = (float*)d_out;

    char* ws = (char*)d_ws;
    const size_t SZ_W1 = (size_t)DD * DD * 2;        // 768x768 bf16
    const size_t SZ_W2 = (size_t)2 * DD * DD * 2;    // 1536x768 bf16
    const size_t SZ_X  = (size_t)MROWS * DD * 2;     // 12544x768 bf16
    unsigned short* wq   = (unsigned short*)(ws);
    unsigned short* wkv  = (unsigned short*)(ws + SZ_W1);
    unsigned short* wpr  = (unsigned short*)(ws + SZ_W1 + SZ_W2);
    unsigned short* xa   = (unsigned short*)(ws + 2 * SZ_W1 + SZ_W2);
    unsigned short* xs   = (unsigned short*)(ws + 2 * SZ_W1 + SZ_W2 + SZ_X);
    unsigned short* qb   = (unsigned short*)(ws + 2 * SZ_W1 + SZ_W2 + 2 * SZ_X);
    unsigned short* kvb  = (unsigned short*)(ws + 2 * SZ_W1 + SZ_W2 + 3 * SZ_X);
    unsigned short* ao   = (unsigned short*)(ws + 2 * SZ_W1 + SZ_W2 + 5 * SZ_X);

    // weights -> bf16
    cvt_bf16_kernel<<<(DD * DD / 4 + 255) / 256, 256, 0, stream>>>(q_w, wq, DD * DD);
    cvt_bf16_kernel<<<(2 * DD * DD / 4 + 255) / 256, 256, 0, stream>>>(kv_w, wkv, 2 * DD * DD);
    cvt_bf16_kernel<<<(DD * DD / 4 + 255) / 256, 256, 0, stream>>>(proj_w, wpr, DD * DD);

    // pos-add + cast (audio: skip 2 rows, s_x: skip 1 row)
    int prep_grid = (int)(((size_t)MROWS * DD / 4 + 255) / 256);
    add_pos_cast_kernel<<<prep_grid, 256, 0, stream>>>(audio + (size_t)2 * 64 * DD,
                                                       audio_sp, audio_tp, xa);
    add_pos_cast_kernel<<<prep_grid, 256, 0, stream>>>(s_x + (size_t)1 * 64 * DD,
                                                       clip_sp, clip_tp, xs);

    // projections
    gemm_bf16_kernel<0><<<dim3(MROWS / 128, DD / 64), 256, 0, stream>>>(
        xa, wq, q_b, qb, MROWS, DD, DD);
    gemm_bf16_kernel<0><<<dim3(MROWS / 128, 2 * DD / 64), 256, 0, stream>>>(
        xs, wkv, kv_b, kvb, MROWS, 2 * DD, DD);

    // attention
    attn_kernel<<<dim3(BB * HH, NTOK / 112), 224, 0, stream>>>(qb, kvb, ao);

    // output projection scattered into final layout + cls rows
    gemm_bf16_kernel<1><<<dim3(MROWS / 128, DD / 64), 256, 0, stream>>>(
        ao, wpr, proj_b, out, MROWS, DD, DD);
    copy_cls_kernel<<<(2 * 64 * DD / 4 + 255) / 256, 256, 0, stream>>>(audio, out);
}
